// XAttn_65214783422648
// MI455X (gfx1250) — matrix operand order repaired and checked by reading, compiled, not run on hardware
//
#include <hip/hip_runtime.h>

typedef __attribute__((ext_vector_type(16))) _Float16 v16h;
typedef __attribute__((ext_vector_type(8)))  _Float16 v8h;
typedef __attribute__((ext_vector_type(4)))  _Float16 v4h;
typedef __attribute__((ext_vector_type(8)))  float    v8f;

#define BATCH 256
#define NA    196
#define NB    196
#define DK    256
#define DIM   1024
#define SMOOTH_F 4.0f

#define NT    13      // 13 N-tiles of 16 -> 208 >= 196
#define NPAD  208
#define APAD  224     // K-dim of GEMM2 padded to 7 chunks of 32
#define ASTR  232     // LDS row stride (halves) of attn[n][a]   (464 B, 16B aligned)
#define KSTR  40      // LDS row stride (halves) of staged K^T   (80 B, 16B aligned)
#define NTHREADS 416  // 13 waves of 32
#define NCHUNK1 (DK / 32)    // 8 k-chunks in phase 1
#define NCHUNK2 (APAD / 32)  // 7 k-chunks in phase 2

// dynamic LDS: attn = NPAD*ASTR halves, stage = 2 * NPAD*KSTR halves  => 129,792 B
#define LDS_BYTES ((NPAD * ASTR + 2 * NPAD * KSTR) * 2)

static __device__ __forceinline__ v16h join8(v8h a, v8h b) {
  return __builtin_shufflevector(a, b, 0,1,2,3,4,5,6,7,8,9,10,11,12,13,14,15);
}

struct Raw16 { float4 a, b, c, d; };

static __device__ __forceinline__ v16h cvt16(const Raw16& r) {
  v16h af;
  af[0]=(_Float16)r.a.x; af[1]=(_Float16)r.a.y; af[2]=(_Float16)r.a.z; af[3]=(_Float16)r.a.w;
  af[4]=(_Float16)r.b.x; af[5]=(_Float16)r.b.y; af[6]=(_Float16)r.b.z; af[7]=(_Float16)r.b.w;
  af[8]=(_Float16)r.c.x; af[9]=(_Float16)r.c.y; af[10]=(_Float16)r.c.z; af[11]=(_Float16)r.c.w;
  af[12]=(_Float16)r.d.x; af[13]=(_Float16)r.d.y; af[14]=(_Float16)r.d.z; af[15]=(_Float16)r.d.w;
  return af;
}

// Q A-fragment raw load: row a = rBase+l2 (guarded), k = c*32 + hi*8 + {0..7,16..23}
static __device__ __forceinline__ Raw16 loadQraw(const float* __restrict__ Qb,
                                                 int rBase, int l2, int hi, int c) {
  Raw16 r = {};
  int a = rBase + l2;
  if (a < NA) {
    const float* p = Qb + (size_t)a * DK + c * 32 + hi * 8;
    r.a = *(const float4*)(p);      r.b = *(const float4*)(p + 4);
    r.c = *(const float4*)(p + 16); r.d = *(const float4*)(p + 20);
  }
  return r;
}

// Va A-fragment raw load: row d (always valid), a = c*32 + hi*8 + {0..7,16..23} (guard a<NA)
static __device__ __forceinline__ Raw16 loadVaRaw(const float* __restrict__ row,
                                                  int hi, int c) {
  Raw16 r = {};
  const int k0 = c * 32 + hi * 8;
  if (k0 + 23 < NA) {
    r.a = *(const float4*)(row + k0);      r.b = *(const float4*)(row + k0 + 4);
    r.c = *(const float4*)(row + k0 + 16); r.d = *(const float4*)(row + k0 + 20);
  } else {
    float v[16];
#pragma unroll
    for (int i = 0; i < 8; ++i) {
      int a0 = k0 + i, a1 = k0 + 16 + i;
      v[i]     = (a0 < NA) ? row[a0] : 0.0f;
      v[8 + i] = (a1 < NA) ? row[a1] : 0.0f;
    }
    r.a = make_float4(v[0], v[1], v[2], v[3]);   r.b = make_float4(v[4], v[5], v[6], v[7]);
    r.c = make_float4(v[8], v[9], v[10], v[11]); r.d = make_float4(v[12], v[13], v[14], v[15]);
  }
  return r;
}

__global__ void __launch_bounds__(NTHREADS)
xattn_fused(const float* __restrict__ Q,   // [B, Na, Dk]
            const float* __restrict__ Va,  // [B, Dim, Na]
            const float* __restrict__ Kb,  // [B, Dk, Nb]
            const float* __restrict__ Vb,  // [B, Dim, Nb]
            const float* __restrict__ gamma,
            float* __restrict__ Out)       // [B, Dim, Nb]
{
  extern __shared__ _Float16 lds[];
  _Float16* attn = lds;                              // [NPAD][ASTR]
  _Float16* st0  = lds + NPAD * ASTR;                // [NPAD][KSTR]
  _Float16* st1  = lds + NPAD * ASTR + NPAD * KSTR;  // [NPAD][KSTR]

  const int b    = blockIdx.x;
  const int tid  = threadIdx.x;
  const int wave = tid >> 5;
  const int lane = tid & 31;
  const int l2   = lane & 15;
  const int hi   = lane >> 4;

  const float g = gamma[0];

  const float* Qb  = Q  + (size_t)b * NA  * DK;
  const float* Kbb = Kb + (size_t)b * DK  * NB;
  const float* Vab = Va + (size_t)b * DIM * NA;
  const float* Vbb = Vb + (size_t)b * DIM * NB;
  float*       Ob  = Out + (size_t)b * DIM * NB;

  // staging unit mapping: 8 k-quads x 208 n = 1664 units = 4 per thread
  int sn[4], skq[4];
#pragma unroll
  for (int i = 0; i < 4; ++i) {
    int idx = tid + i * NTHREADS;
    sn[i]  = idx % NPAD;
    skq[i] = idx / NPAD;  // 0..7
  }

  // raw k-chunk staging loads (issued early; converted+stored after WMMAs)
  float sr[16];
  auto stageLoad = [&](int c) {
#pragma unroll
    for (int i = 0; i < 4; ++i) {
      const bool ok = sn[i] < NB;
      const size_t base = (size_t)(c * 32 + skq[i] * 4) * NB + sn[i];
      sr[4*i+0] = ok ? Kbb[base]          : 0.0f;
      sr[4*i+1] = ok ? Kbb[base + NB]     : 0.0f;
      sr[4*i+2] = ok ? Kbb[base + 2*NB]   : 0.0f;
      sr[4*i+3] = ok ? Kbb[base + 3*NB]   : 0.0f;
    }
  };
  auto stageStore = [&](_Float16* buf) {
#pragma unroll
    for (int i = 0; i < 4; ++i) {
      v4h h;
      h[0] = (_Float16)sr[4*i+0]; h[1] = (_Float16)sr[4*i+1];
      h[2] = (_Float16)sr[4*i+2]; h[3] = (_Float16)sr[4*i+3];
      *(v4h*)(buf + sn[i] * KSTR + skq[i] * 4) = h;  // ds_store_b64
    }
  };

  // ================= Phase 1: attn = softmax(SMOOTH * Q K) =================
  v8f acc[NT];
#pragma unroll
  for (int t = 0; t < NT; ++t) acc[t] = (v8f){};

  const int rBase = wave * 16;  // this wave's 16 Q rows (a dimension)

  stageLoad(0);
  stageStore(st0);
  v16h af = cvt16(loadQraw(Qb, rBase, l2, hi, 0));
  __syncthreads();

  for (int c = 0; c < NCHUNK1; ++c) {
    _Float16* cur = (c & 1) ? st1 : st0;
    _Float16* nxt = (c & 1) ? st0 : st1;

    // issue next chunk's global loads before consuming this chunk
    Raw16 rq = {};
    if (c < NCHUNK1 - 1) {
      stageLoad(c + 1);
      rq = loadQraw(Qb, rBase, l2, hi, c + 1);
    }

    // pipelined B-fragment loads + WMMA over 13 N-tiles
    v8h b0, b1;
    {
      const _Float16* p = cur + l2 * KSTR + hi * 8;
      b0 = *(const v8h*)p; b1 = *(const v8h*)(p + 16);
    }
#pragma unroll
    for (int t = 0; t < NT; ++t) {
      v16h bf = join8(b0, b1);
      if (t < NT - 1) {
        const _Float16* p = cur + ((t + 1) * 16 + l2) * KSTR + hi * 8;
        b0 = *(const v8h*)p; b1 = *(const v8h*)(p + 16);
      }
      acc[t] = __builtin_amdgcn_wmma_f32_16x16x32_f16(
          false, af, false, bf, (short)0, acc[t], false, false);
    }

    if (c < NCHUNK1 - 1) {
      af = cvt16(rq);     // waits for Q loads here (behind WMMAs)
      stageStore(nxt);    // waits for K loads here (behind WMMAs)
      __syncthreads();
    }
  }

  // ---- softmax over n (rows stripe as M = rBase + hi*8 + j) ----
  float mx[8], sm[8], rcp[8];
#pragma unroll
  for (int j = 0; j < 8; ++j) mx[j] = -3.0e38f;
#pragma unroll
  for (int t = 0; t < NT; ++t) {
#pragma unroll
    for (int j = 0; j < 8; ++j) {
      float v = acc[t][j] * SMOOTH_F;
      if (t == NT - 1 && l2 >= (NB - (NT - 1) * 16)) v = -1.0e30f;  // n >= 196
      acc[t][j] = v;
      mx[j] = fmaxf(mx[j], v);
    }
  }
#pragma unroll
  for (int j = 0; j < 8; ++j) {
#pragma unroll
    for (int off = 8; off >= 1; off >>= 1)
      mx[j] = fmaxf(mx[j], __shfl_xor(mx[j], off, 32));
    sm[j] = 0.0f;
  }
#pragma unroll
  for (int t = 0; t < NT; ++t) {
#pragma unroll
    for (int j = 0; j < 8; ++j) {
      float e = __expf(acc[t][j] - mx[j]);
      acc[t][j] = e;
      sm[j] += e;
    }
  }
#pragma unroll
  for (int j = 0; j < 8; ++j) {
#pragma unroll
    for (int off = 8; off >= 1; off >>= 1)
      sm[j] += __shfl_xor(sm[j], off, 32);
    rcp[j] = 1.0f / sm[j];
  }

  // store attn[n][a] (f16), zeroing rows a >= 196
#pragma unroll
  for (int t = 0; t < NT; ++t) {
    int n = t * 16 + l2;
    v8h h;
#pragma unroll
    for (int j = 0; j < 8; ++j) {
      int a = rBase + hi * 8 + j;
      float v = (a < NA) ? acc[t][j] * rcp[j] : 0.0f;
      h[j] = (_Float16)v;
    }
    *(v8h*)(attn + n * ASTR + rBase + hi * 8) = h;
  }
  // zero-fill a in [208, 224): one 16B store per thread covers 208 rows x 16 halves
  {
    int n = tid >> 1, part = tid & 1;
    *(v8h*)(attn + n * ASTR + NPAD + part * 8) = (v8h){};
  }
  __syncthreads();

  // ================= Phase 2: Out = Va * attn + Vb + gamma =================
  for (int mt = wave; mt < DIM / 16; mt += NT) {
    const int dBase = mt * 16;
    const float* row = Vab + (size_t)(dBase + l2) * NA;

    v8f oacc[NT];
#pragma unroll
    for (int t = 0; t < NT; ++t) oacc[t] = (v8f){};

    v16h af2 = cvt16(loadVaRaw(row, hi, 0));
    v8h b0, b1;
    {
      const _Float16* p = attn + l2 * ASTR + hi * 8;
      b0 = *(const v8h*)p; b1 = *(const v8h*)(p + 16);
    }

    for (int c = 0; c < NCHUNK2; ++c) {
      Raw16 rw = {};
      if (c < NCHUNK2 - 1) rw = loadVaRaw(row, hi, c + 1);  // prefetch next A chunk

#pragma unroll
      for (int t = 0; t < NT; ++t) {
        v16h bf = join8(b0, b1);
        // prefetch next fragment (crosses chunk boundary; attn is stable)
        if (!(c == NCHUNK2 - 1 && t == NT - 1)) {
          const int tn = (t == NT - 1) ? 0 : t + 1;
          const int cn = (t == NT - 1) ? c + 1 : c;
          const _Float16* p = attn + (tn * 16 + l2) * ASTR + cn * 32 + hi * 8;
          b0 = *(const v8h*)p; b1 = *(const v8h*)(p + 16);
        }
        oacc[t] = __builtin_amdgcn_wmma_f32_16x16x32_f16(
            false, af2, false, bf, (short)0, oacc[t], false, false);
      }
      if (c < NCHUNK2 - 1) af2 = cvt16(rw);  // conversion (and its wait) behind WMMAs
    }

    // epilogue: + value_b + gamma, store fp32
#pragma unroll
    for (int t = 0; t < NT; ++t) {
      int n = t * 16 + l2;
      if (n < NB) {
#pragma unroll
        for (int j = 0; j < 8; ++j) {
          int d = dBase + hi * 8 + j;
          size_t off = (size_t)d * NB + n;
          Ob[off] = oacc[t][j] + Vbb[off] + g;
        }
      }
    }
  }
}

extern "C" void kernel_launch(void* const* d_in, const int* in_sizes, int n_in,
                              void* d_out, int out_size, void* d_ws, size_t ws_size,
                              hipStream_t stream) {
  (void)in_sizes; (void)n_in; (void)d_ws; (void)ws_size; (void)out_size;
  const float* Q  = (const float*)d_in[0];
  const float* Va = (const float*)d_in[1];
  const float* Kb = (const float*)d_in[2];
  const float* Vb = (const float*)d_in[3];
  const float* gm = (const float*)d_in[4];
  float* Out = (float*)d_out;

  (void)hipFuncSetAttribute((const void*)xattn_fused,
                            hipFuncAttributeMaxDynamicSharedMemorySize,
                            LDS_BYTES);
  xattn_fused<<<dim3(BATCH), dim3(NTHREADS), LDS_BYTES, stream>>>(
      Q, Va, Kb, Vb, gm, Out);
}
